// EdgeClassifier_10496900071609
// MI455X (gfx1250) — matrix operand order repaired and checked by reading, not compiled, not run on hardware
//
#include <hip/hip_runtime.h>

typedef __attribute__((ext_vector_type(16))) _Float16 v16h;
typedef __attribute__((ext_vector_type(8)))  _Float16 v8h;
typedef __attribute__((ext_vector_type(8)))  float    v8f;

#define HD   128            // hidden dim / W1 rows
#define KD   256            // concat dim / W1 cols
#define LDSW 264            // 256 + 8 halves pad -> avoids LDS bank conflicts across n

// One wave handles 32 edges x 128 outputs. K=256 in 8 steps of 32.
// A (16x32 f16): lane L holds edge m=L%16; elem j -> k = 8*(L/16) + (j<8 ? j : j+8).
// B (32x16 f16): lane L holds col  n=L%16; elem j -> k = 8*(L/16) + (j<8 ? j : j+8).
// C/D (16x16 f32): lane L col n=L%16; vreg r -> row m = r + 8*(L/16).
__global__ __launch_bounds__(256, 4) void edge_mlp_wmma_f16(
    const float* __restrict__ z_drug,
    const float* __restrict__ z_prot,
    const int*   __restrict__ eli,      // [2,E] row-major: row then col
    const float* __restrict__ W1,       // [128,256]
    const float* __restrict__ b1,       // [128]
    const float* __restrict__ W2,       // [128]
    const float* __restrict__ b2,       // [1]
    float* __restrict__ out,            // [E]
    int E, int num_tiles)               // tiles of 32 edges
{
    __shared__ __align__(16) _Float16 w1lds[HD * LDSW];   // 67,584 B

    const int tid = threadIdx.x;

    // ---- stage W1 (f32 -> f16) into LDS, row n padded to LDSW halves ----
    for (int idx = tid * 4; idx < HD * KD; idx += 256 * 4) {
        const int n = idx >> 8;          // /256
        const int k = idx & 255;
        float4 w = *(const float4*)(W1 + idx);
        _Float16* dst = &w1lds[n * LDSW + k];
        dst[0] = (_Float16)w.x; dst[1] = (_Float16)w.y;
        dst[2] = (_Float16)w.z; dst[3] = (_Float16)w.w;
    }
    __syncthreads();

    const int lane = tid & 31;
    const int wave = tid >> 5;
    const int mcol = lane & 15;          // A: edge-in-subtile / C,D & B: column n (mod 16)
    const int hb   = lane >> 4;          // lane half

    // per-lane epilogue constants for n = mcol + 16*t
    float b1v[8], w2v[8];
    #pragma unroll
    for (int t = 0; t < 8; ++t) {
        const int n = mcol + 16 * t;
        b1v[t] = b1[n];
        w2v[t] = W2[n];
    }
    const float bias2 = b2[0];

    const int wave_stride = gridDim.x * 8;
    for (int tile = blockIdx.x * 8 + wave; tile < num_tiles; tile += wave_stride) {
        // Opaque memory barrier: stops LICM from hoisting the 64 loop-invariant
        // B-fragment LDS loads out of the tile loop (which ballooned to ~512
        // VGPRs + scratch spills). B is cheap to re-read from LDS every tile.
        asm volatile("" ::: "memory");

        const int e0 = tile * 32;

        // gathered base pointers for the two 16-edge sub-tiles
        const float* pd[2];
        const float* pp[2];
        #pragma unroll
        for (int s = 0; s < 2; ++s) {
            int e = e0 + 16 * s + mcol;
            if (e >= E) e = E - 1;
            pd[s] = z_drug + (size_t)eli[e]      * HD;
            pp[s] = z_prot + (size_t)eli[E + e]  * HD;
        }

        v8f acc[2][8];
        #pragma unroll
        for (int s = 0; s < 2; ++s)
            #pragma unroll
            for (int t = 0; t < 8; ++t) acc[s][t] = {};

        #pragma unroll
        for (int kb = 0; kb < 8; ++kb) {
            const int k0 = kb * 32;
            v16h A[2];
            #pragma unroll
            for (int s = 0; s < 2; ++s) {
                const float* src = (k0 < HD) ? (pd[s] + k0) : (pp[s] + (k0 - HD));
                float4 a0 = *(const float4*)(src + 8 * hb);
                float4 a1 = *(const float4*)(src + 8 * hb + 4);
                float4 a2 = *(const float4*)(src + 16 + 8 * hb);
                float4 a3 = *(const float4*)(src + 16 + 8 * hb + 4);
                A[s][0]  = (_Float16)a0.x; A[s][1]  = (_Float16)a0.y;
                A[s][2]  = (_Float16)a0.z; A[s][3]  = (_Float16)a0.w;
                A[s][4]  = (_Float16)a1.x; A[s][5]  = (_Float16)a1.y;
                A[s][6]  = (_Float16)a1.z; A[s][7]  = (_Float16)a1.w;
                A[s][8]  = (_Float16)a2.x; A[s][9]  = (_Float16)a2.y;
                A[s][10] = (_Float16)a2.z; A[s][11] = (_Float16)a2.w;
                A[s][12] = (_Float16)a3.x; A[s][13] = (_Float16)a3.y;
                A[s][14] = (_Float16)a3.z; A[s][15] = (_Float16)a3.w;
            }

            // Batch-issue all 8 B-fragment LDS loads for this K-step first so
            // the WMMAs below can retire with progressive s_wait_dscnt (latency
            // of later fragments hidden under earlier WMMAs).
            v16h Bf[8];
            #pragma unroll
            for (int t = 0; t < 8; ++t) {
                const _Float16* bp = &w1lds[(mcol + 16 * t) * LDSW + k0 + 8 * hb];
                v8h blo = *(const v8h*)bp;
                v8h bhi = *(const v8h*)(bp + 16);
                Bf[t] = __builtin_shufflevector(blo, bhi,
                          0,1,2,3,4,5,6,7,8,9,10,11,12,13,14,15);
            }
            #pragma unroll
            for (int t = 0; t < 8; ++t) {
                acc[0][t] = __builtin_amdgcn_wmma_f32_16x16x32_f16(
                    false, A[0], false, Bf[t], (short)0, acc[0][t], false, false);
                acc[1][t] = __builtin_amdgcn_wmma_f32_16x16x32_f16(
                    false, A[1], false, Bf[t], (short)0, acc[1][t], false, false);
            }
        }

        // ---- epilogue: +b1, relu, dot with W2 over n, +b2 ----
        #pragma unroll
        for (int s = 0; s < 2; ++s) {
            float part[8];
            #pragma unroll
            for (int r = 0; r < 8; ++r) part[r] = 0.0f;
            #pragma unroll
            for (int t = 0; t < 8; ++t) {
                #pragma unroll
                for (int r = 0; r < 8; ++r) {
                    float hv = acc[s][t][r] + b1v[t];
                    hv = hv > 0.0f ? hv : 0.0f;
                    part[r] += hv * w2v[t];
                }
            }
            // reduce across the 16 lanes of each half (wave32 XOR shuffles)
            #pragma unroll
            for (int msk = 1; msk <= 8; msk <<= 1) {
                #pragma unroll
                for (int r = 0; r < 8; ++r)
                    part[r] += __shfl_xor(part[r], msk, 32);
            }
            if (mcol == 0) {
                #pragma unroll
                for (int r = 0; r < 8; ++r) {
                    const int eo = e0 + 16 * s + 8 * hb + r;
                    if (eo < E) out[eo] = part[r] + bias2;
                }
            }
        }
    }
}

extern "C" void kernel_launch(void* const* d_in, const int* in_sizes, int n_in,
                              void* d_out, int out_size, void* d_ws, size_t ws_size,
                              hipStream_t stream) {
    const float* z_drug = (const float*)d_in[0];
    const float* z_prot = (const float*)d_in[1];
    const int*   eli    = (const int*)  d_in[2];
    const float* W1     = (const float*)d_in[3];
    const float* b1     = (const float*)d_in[4];
    const float* W2     = (const float*)d_in[5];
    const float* b2     = (const float*)d_in[6];
    float*       out    = (float*)d_out;

    const int E = in_sizes[2] / 2;                 // edge_label_index is [2,E]
    const int num_tiles = (E + 31) / 32;           // 32 edges per wave-tile
    int blocks = (num_tiles + 7) / 8;              // 8 waves per 256-thread block
    if (blocks > 2048) blocks = 2048;              // grid-stride covers the rest

    hipLaunchKernelGGL(edge_mlp_wmma_f16, dim3(blocks), dim3(256), 0, stream,
                       z_drug, z_prot, eli, W1, b1, W2, b2, out, E, num_tiles);
}
